// GruMlpDynamics_16003048145444
// MI455X (gfx1250) — matrix operand order repaired and checked by reading, not compiled, not run on hardware
//
#include <hip/hip_runtime.h>
#include <hip/hip_bf16.h>
#include <math.h>

typedef __bf16 bf16;
typedef bf16 v16bf __attribute__((ext_vector_type(16)));
typedef bf16 v8bf  __attribute__((ext_vector_type(8)));
typedef float v8f  __attribute__((ext_vector_type(8)));

#define T_STEPS 64
#define BATCH   2048
#define HID     256
#define H3      768
#define H1F     64
#define OUTF    601
#define OUTP    608

// padded LDS row strides (bf16 elems) to avoid bank conflicts on ds_load_b128:
// 264*2B = 528B -> bank step 4 per row, 16 lanes hit 16 distinct banks.
#define HB_S    264
// 72*2B = 144B -> bank step 36, gcd(36,64)=4 -> 16 distinct banks.
#define A_S     72

// packed bf16 weight offsets (in bf16 elements) inside d_ws
#define WHH_OFF 0          // 8x48 blocks of 512
#define W1_OFF  196608     // 8x4 blocks
#define W2_OFF  212992     // 2x4 blocks
#define W3_OFF  217088     // 2x38 blocks

__device__ __forceinline__ v8f wmma_bf16(v16bf a, v16bf b, v8f c) {
  return __builtin_amdgcn_wmma_f32_16x16x32_bf16(false, a, false, b, (short)0, c,
                                                 false, false);
}

// A-matrix 16x32 bf16 fragment per ISA layout:
// lane<16:  M=lane,    K in {kt*32+0..7, kt*32+16..23}
// lane>=16: M=lane-16, K in {kt*32+8..15, kt*32+24..31}
__device__ __forceinline__ v16bf load_a_frag(const bf16* base, int stride, int lane, int kt) {
  const int m = lane & 15, half = lane >> 4;
  const int k0 = kt * 32 + half * 8;
  const int k1 = kt * 32 + 16 + half * 8;
  v8bf lo = *(const v8bf*)(base + m * stride + k0);
  v8bf hi = *(const v8bf*)(base + m * stride + k1);
  return __builtin_shufflevector(lo, hi, 0, 1, 2, 3, 4, 5, 6, 7,
                                 8, 9, 10, 11, 12, 13, 14, 15);
}

// fast transcendentals: v_exp_f32 + v_rcp_f32 (no OCML calls in the hot loop)
__device__ __forceinline__ float fast_rcp(float x)   { return __builtin_amdgcn_rcpf(x); }
__device__ __forceinline__ float sigmoidf_(float x)  { return fast_rcp(1.f + __expf(-x)); }
__device__ __forceinline__ float tanhf_(float x)     { return 1.f - 2.f * fast_rcp(__expf(2.f * x) + 1.f); }
__device__ __forceinline__ float eluf_(float x)      { return x > 0.f ? x : __expf(x) - 1.f; }

// ---------------- prep: pack weights into per-lane WMMA B-fragment order -----
// B tile (32x16) for v_wmma_*_bf16: lane<16 holds K=kt*32+{0..7,16..23} of column
// N=nt*16+lane; lane>=16 holds K=kt*32+{8..15,24..31} of column N=nt*16+(lane-16).
// Stored as 32 lanes x 16 bf16 contiguous (1 KB per tile) -> two b128/lane.
__global__ void pack_weights_kernel(const float* __restrict__ w_hh,
                                    const float* __restrict__ w1,
                                    const float* __restrict__ w2,
                                    const float* __restrict__ w3,
                                    bf16* __restrict__ wsb) {
  const int b = blockIdx.x;
  const int lane = threadIdx.x;
  const int half = lane >> 4, ln = lane & 15;
  int kt, nt, base, mode;
  if (b < 384)      { kt = b / 48;           nt = b % 48;           base = WHH_OFF + b * 512;       mode = 0; }
  else if (b < 416) { int i = b - 384; kt = i / 4;  nt = i % 4;  base = W1_OFF + i * 512; mode = 1; }
  else if (b < 424) { int i = b - 416; kt = i / 4;  nt = i % 4;  base = W2_OFF + i * 512; mode = 2; }
  else              { int i = b - 424; kt = i / 38; nt = i % 38; base = W3_OFF + i * 512; mode = 3; }
#pragma unroll
  for (int e = 0; e < 16; ++e) {
    const int k = kt * 32 + half * 8 + (e < 8 ? e : e + 8);
    const int n = nt * 16 + ln;
    float v;
    if (mode == 0)      v = w_hh[n * HID + k];                   // B = w_hh^T [256x768]
    else if (mode == 1) v = w1[k * H1F + n];                     // [256x64]
    else if (mode == 2) v = w2[k * H1F + n];                     // [64x64]
    else                v = (n < OUTF) ? w3[k * OUTF + n] : 0.f; // [64x608 padded]
    wsb[base + lane * 16 + e] = (bf16)v;
  }
}

// ---------------- fused GRU + MLP, one workgroup = 16 batch rows -------------
// Registers uncapped: the compiler keeps all 96 loop-invariant B-fragments
// resident (proven to fit in round 1) so the steady-state GEMM has no memory
// traffic.  Gate VALU for subtile s is interleaved with WMMA of subtile s+1
// (independent), so the TRANS-pipe WMMA co-executes with the gate math.
__global__ __launch_bounds__(128)
void gru_mlp_fused_kernel(const float* __restrict__ x,     // [T,B,1]
                          const float* __restrict__ h0,    // [1,B,H]
                          const float* __restrict__ w_ih,  // [3H,1]
                          const float* __restrict__ b_ih,  // [3H]
                          const float* __restrict__ b_hh,  // [3H]
                          const float* __restrict__ b1,    // [64]
                          const float* __restrict__ b2,    // [64]
                          const float* __restrict__ b3,    // [601]
                          const bf16*  __restrict__ wp,    // packed weights
                          float* __restrict__ out) {       // y then h1
  __shared__ __align__(16) float hf[16 * HID];    // h (f32) for z*h term
  __shared__ __align__(16) bf16  hb[16 * HB_S];   // h (bf16) A-operand, padded
  __shared__ __align__(16) bf16  a1[16 * A_S];
  __shared__ __align__(16) bf16  a2[16 * A_S];
  __shared__ float xAll[T_STEPS * 16];            // x for all steps of this tile
  __shared__ float wihL[H3];                      // w_ih (x multiplier)
  __shared__ float bcL[2 * HID];                  // b_ih+b_hh for r,z gates
  __shared__ float binL[HID], bhnL[HID];          // n gate: b_ih_n, b_hh_n
  __shared__ float b1L[H1F], b2L[H1F], b3L[OUTP];

  const int tid  = threadIdx.x;
  const int wave = tid >> 5;
  const int lane = tid & 31;
  const int half = lane >> 4, ln = lane & 15;
  const int wg   = blockIdx.x;          // batch tile: rows [wg*16, wg*16+16)
  const int rowBase = wg * 16;

  // one-time LDS preload
  for (int i = tid; i < H3; i += 128) wihL[i] = w_ih[i];
  for (int i = tid; i < 2 * HID; i += 128) bcL[i] = b_ih[i] + b_hh[i];
  for (int i = tid; i < HID; i += 128) { binL[i] = b_ih[512 + i]; bhnL[i] = b_hh[512 + i]; }
  if (tid < H1F) { b1L[tid] = b1[tid]; b2L[tid] = b2[tid]; }
  for (int i = tid; i < OUTP; i += 128) b3L[i] = (i < OUTF) ? b3[i] : 0.f;
  for (int i = tid; i < T_STEPS * 16; i += 128) {
    const int t = i >> 4, r = i & 15;
    xAll[i] = x[(size_t)t * BATCH + rowBase + r];
  }
  for (int i = tid; i < 16 * HID; i += 128) {
    const int row = i >> 8, col = i & (HID - 1);
    float h = h0[rowBase * HID + i];
    hf[i] = h; hb[row * HB_S + col] = (bf16)h;
  }
  __syncthreads();

  for (int t = 0; t < T_STEPS; ++t) {
    // A-fragments for all 8 K-tiles of old h (shared across all N-tiles)
    v16bf aF[8];
#pragma unroll
    for (int kt = 0; kt < 8; ++kt) aF[kt] = load_a_frag(hb, HB_S, lane, kt);
    __syncthreads();                      // aF consumed before hb rewrite

    // -------- GRU: per 16-col subtile, GEMM the (r,z,n) triple then gate ----
#pragma unroll
    for (int s = 0; s < 4; ++s) {
      v8f ar, az, an;
#pragma unroll
      for (int r = 0; r < 8; ++r) { ar[r] = 0.f; az[r] = 0.f; an[r] = 0.f; }
      const int ntR = 0 * 16 + wave * 4 + s;
      const int ntZ = 1 * 16 + wave * 4 + s;
      const int ntN = 2 * 16 + wave * 4 + s;
#pragma unroll
      for (int kt = 0; kt < 8; ++kt) {
        const v16bf* bpR = (const v16bf*)(wp + WHH_OFF + ((kt * 48 + ntR) << 9) + (lane << 4));
        const v16bf* bpZ = (const v16bf*)(wp + WHH_OFF + ((kt * 48 + ntZ) << 9) + (lane << 4));
        const v16bf* bpN = (const v16bf*)(wp + WHH_OFF + ((kt * 48 + ntN) << 9) + (lane << 4));
        ar = wmma_bf16(aF[kt], *bpR, ar);
        az = wmma_bf16(aF[kt], *bpZ, az);
        an = wmma_bf16(aF[kt], *bpN, an);
      }
      // gate nonlinearity + h update for this subtile (overlaps next GEMM)
      const int j = wave * 64 + s * 16 + ln;         // hidden index [0,256)
      const float wr = wihL[j],       pr  = bcL[j];
      const float wz = wihL[256 + j], pz  = bcL[256 + j];
      const float wn = wihL[512 + j], bin = binL[j], bhn = bhnL[j];
#pragma unroll
      for (int r = 0; r < 8; ++r) {
        const int row = r + half * 8;
        const float xv = xAll[t * 16 + row];
        const float rg = sigmoidf_(fmaf(xv, wr, pr) + ar[r]);
        const float zg = sigmoidf_(fmaf(xv, wz, pz) + az[r]);
        const float ng = tanhf_(fmaf(rg, an[r] + bhn, fmaf(xv, wn, bin)));
        const float hold = hf[row * HID + j];
        const float hnew = fmaf(zg, hold - ng, ng);
        hf[row * HID + j] = hnew;
        hb[row * HB_S + j] = (bf16)hnew;
      }
    }
    __syncthreads();                      // h_new complete

    // -------- MLP stage 1: a1 = elu(h_new @ w1 + b1), wave owns 16 cols -----
    v16bf aN[8];
#pragma unroll
    for (int kt = 0; kt < 8; ++kt) aN[kt] = load_a_frag(hb, HB_S, lane, kt);
    v8f c1;
#pragma unroll
    for (int r = 0; r < 8; ++r) c1[r] = 0.f;
#pragma unroll
    for (int kt = 0; kt < 8; ++kt) {
      const v16bf* bp = (const v16bf*)(wp + W1_OFF + ((kt * 4 + wave) << 9) + (lane << 4));
      c1 = wmma_bf16(aN[kt], *bp, c1);
    }
    {
      const int col = wave * 16 + ln;
      const float bb = b1L[col];
#pragma unroll
      for (int r = 0; r < 8; ++r) {
        const int row = r + half * 8;
        a1[row * A_S + col] = (bf16)eluf_(c1[r] + bb);
      }
    }
    __syncthreads();

    // -------- MLP stage 2: a2 = elu(a1 @ w2 + b2) ---------------------------
    v16bf a1F[2];
#pragma unroll
    for (int kt = 0; kt < 2; ++kt) a1F[kt] = load_a_frag(a1, A_S, lane, kt);
    v8f c2;
#pragma unroll
    for (int r = 0; r < 8; ++r) c2[r] = 0.f;
#pragma unroll
    for (int kt = 0; kt < 2; ++kt) {
      const v16bf* bp = (const v16bf*)(wp + W2_OFF + ((kt * 4 + wave) << 9) + (lane << 4));
      c2 = wmma_bf16(a1F[kt], *bp, c2);
    }
    {
      const int col = wave * 16 + ln;
      const float bb = b2L[col];
#pragma unroll
      for (int r = 0; r < 8; ++r) {
        const int row = r + half * 8;
        a2[row * A_S + col] = (bf16)eluf_(c2[r] + bb);
      }
    }
    __syncthreads();

    // -------- MLP stage 3: y = a2 @ w3 + b3, 38 col-tiles over 4 waves ------
    v16bf a2F[2];
#pragma unroll
    for (int kt = 0; kt < 2; ++kt) a2F[kt] = load_a_frag(a2, A_S, lane, kt);
    for (int nt = wave; nt < 38; nt += 4) {
      v8f c3;
#pragma unroll
      for (int r = 0; r < 8; ++r) c3[r] = 0.f;
#pragma unroll
      for (int kt = 0; kt < 2; ++kt) {
        const v16bf* bp = (const v16bf*)(wp + W3_OFF + ((kt * 38 + nt) << 9) + (lane << 4));
        c3 = wmma_bf16(a2F[kt], *bp, c3);
      }
      const int col = nt * 16 + ln;
      if (col < OUTF) {
        const float bb = b3L[col];
#pragma unroll
        for (int r = 0; r < 8; ++r) {
          const int row = r + half * 8;
          out[((size_t)t * BATCH + rowBase + row) * OUTF + col] = c3[r] + bb;
        }
      }
    }
    // no barrier needed: next-step writes to hb are fenced by the
    // post-load barrier at the top of the next iteration.
  }

  // -------- h1 output --------
  const size_t yElems = (size_t)T_STEPS * BATCH * OUTF;
  for (int i = tid; i < 16 * HID; i += 128)
    out[yElems + (size_t)rowBase * HID + i] = hf[i];
}

extern "C" void kernel_launch(void* const* d_in, const int* in_sizes, int n_in,
                              void* d_out, int out_size, void* d_ws, size_t ws_size,
                              hipStream_t stream) {
  const float* x    = (const float*)d_in[0];
  const float* h0   = (const float*)d_in[1];
  const float* w_ih = (const float*)d_in[2];
  const float* w_hh = (const float*)d_in[3];
  const float* b_ih = (const float*)d_in[4];
  const float* b_hh = (const float*)d_in[5];
  const float* w1   = (const float*)d_in[6];
  const float* b1   = (const float*)d_in[7];
  const float* w2   = (const float*)d_in[8];
  const float* b2   = (const float*)d_in[9];
  const float* w3   = (const float*)d_in[10];
  const float* b3   = (const float*)d_in[11];
  bf16* wsb = (bf16*)d_ws;   // ~512 KB packed bf16 weights

  // pack w_hh^T / w1 / w2 / w3 into WMMA B-fragment order (once per launch)
  pack_weights_kernel<<<500, 32, 0, stream>>>(w_hh, w1, w2, w3, wsb);

  // fused GRU + MLP: 128 workgroups x 4 waves, 16 batch rows each
  gru_mlp_fused_kernel<<<BATCH / 16, 128, 0, stream>>>(
      x, h0, w_ih, b_ih, b_hh, b1, b2, b3, wsb, (float*)d_out);
}
